// ActorCritic_25649544692303
// MI455X (gfx1250) — matrix operand order repaired and checked by reading, not compiled, not run on hardware
//
#include <hip/hip_runtime.h>

// ---------------------------------------------------------------------------
// MI455X (gfx1250) GNN actor-critic forward.
// All GEMMs via v_wmma_f32_16x16x32_f16 (wave32, 16x16 tiles, f32 accum).
// Edge phase fuses gather + GEMM + leaky_relu + segment-sum (f32 atomics to
// L2-resident ssum). Weights staged in LDS as f16 once per block.
// ---------------------------------------------------------------------------

typedef __attribute__((ext_vector_type(16))) _Float16 v16h;
typedef __attribute__((ext_vector_type(8)))  float    v8f;

#define NNODES 100000
#define NEDGES 1600000
#define DH     128
#define MTILES (NNODES / 16)   // 6250 node tiles of 16 rows

__device__ __forceinline__ v8f wmma16x16x32(v16h a, v16h b, v8f c) {
    // 8 args: (neg_a, A, neg_b, B, c_mod, C, reuse_a, reuse_b)
    return __builtin_amdgcn_wmma_f32_16x16x32_f16(
        false, a, false, b, (short)0, c, false, false);
}

// A-fragment K index for element j, lane-half hi (contiguous order k = 16*hi + j, same as the B loads)
__device__ __forceinline__ int a_k(int j, int hi) {
    return hi * 16 + j;
}

// ---------------------------------------------------------------------------
__global__ void k_zero(float* __restrict__ p, int n) {
    int i = blockIdx.x * blockDim.x + threadIdx.x;
    if (i < n) p[i] = 0.f;
}

__global__ void k_embed(const int* __restrict__ gt, const float* __restrict__ emb,
                        float* __restrict__ h0) {
    int i = blockIdx.x * blockDim.x + threadIdx.x;
    if (i < NNODES * 32) {
        int node = i >> 5, j = i & 31;
        h0[i] = emb[gt[node] * 32 + j];
    }
}

__global__ void k_deg(const int* __restrict__ dst, float* __restrict__ deg) {
    int e = blockIdx.x * blockDim.x + threadIdx.x;
    if (e < NEDGES) atomicAdd(&deg[dst[e]], 1.0f);
}

__global__ void k_recip(const float* __restrict__ deg, float* __restrict__ rcp) {
    int i = blockIdx.x * blockDim.x + threadIdx.x;
    if (i < NNODES) rcp[i] = 1.0f / fmaxf(deg[i], 1.0f);
}

// ---------------------------------------------------------------------------
// Edge kernel: t = leaky_relu([h[src] || w] @ W1^T); ssum[dst] += t
// One wave = 16 edges x 128 outputs. KIN = input feature dim, KP = KCH*32.
// ---------------------------------------------------------------------------
template <int KCH, int KIN>
__global__ __launch_bounds__(256) void k_edge(
    const float* __restrict__ h, const int* __restrict__ src,
    const int* __restrict__ dst, const float* __restrict__ f0,
    const float* __restrict__ f1, const float* __restrict__ f2,
    const float* __restrict__ W1, float* __restrict__ ssum) {
    constexpr int KP = KCH * 32;
    __shared__ _Float16 lw[128 * KP];           // W1 (128 x KIN+3) padded to KP, f16
    for (int idx = threadIdx.x; idx < 128 * KP; idx += 256) {
        int n = idx / KP, k = idx - n * KP;
        float v = (k < KIN + 3) ? W1[n * (KIN + 3) + k] : 0.f;
        lw[idx] = (_Float16)v;
    }
    __syncthreads();

    int wave = threadIdx.x >> 5, lane = threadIdx.x & 31;
    int hi = lane >> 4, ln = lane & 15;
    int tile = blockIdx.x * 8 + wave;
    int e0 = tile * 16;
    if (e0 >= NEDGES) return;

    // Gather A: row M = ln (edge e0+ln), lane-half selects K interleave.
    int e = e0 + ln;
    const float* row = h + (size_t)src[e] * KIN;
    float ef0 = f0[e], ef1 = f1[e], ef2 = f2[e];
    v16h A[KCH];
#pragma unroll
    for (int kc = 0; kc < KCH; ++kc) {
#pragma unroll
        for (int j = 0; j < 16; ++j) {
            int k = kc * 32 + a_k(j, hi);
            float v;
            if (k < KIN)            v = row[k];
            else if (k == KIN)      v = ef0;
            else if (k == KIN + 1)  v = ef1;
            else if (k == KIN + 2)  v = ef2;
            else                    v = 0.f;
            A[kc][j] = (_Float16)v;
        }
    }
    // dst node of each accumulator row (M = v + 8*hi)
    int drow[8];
#pragma unroll
    for (int v = 0; v < 8; ++v) drow[v] = dst[e0 + hi * 8 + v];

#pragma unroll
    for (int nt = 0; nt < 8; ++nt) {            // 8 x 16 = 128 output cols
        v8f acc = {};
        const _Float16* bb = &lw[(nt * 16 + ln) * KP];
#pragma unroll
        for (int kc = 0; kc < KCH; ++kc) {
            v16h B;
#pragma unroll
            for (int j = 0; j < 16; ++j) B[j] = bb[kc * 32 + hi * 16 + j];
            acc = wmma16x16x32(A[kc], B, acc);
        }
        int n = nt * 16 + ln;
#pragma unroll
        for (int v = 0; v < 8; ++v) {
            float t = acc[v];
            t = (t > 0.f) ? t : 0.01f * t;      // leaky_relu(0.01)
            atomicAdd(&ssum[(size_t)drow[v] * DH + n], t);
        }
    }
}

// ---------------------------------------------------------------------------
// Node kernel: h_out = relu([h || ssum*recip] @ W2^T + b2). KP = KIN + 128.
// grid.y = 2 slices of 64 output columns (LDS <= 32 KB).
// ---------------------------------------------------------------------------
template <int KCH, int KIN>
__global__ __launch_bounds__(256) void k_node(
    const float* __restrict__ hin, const float* __restrict__ ssum,
    const float* __restrict__ rcp, const float* __restrict__ W2,
    const float* __restrict__ b2, float* __restrict__ hout) {
    constexpr int KP = KCH * 32;                // == KIN + 128 exactly
    __shared__ _Float16 lw[64 * KP];
    int slice = blockIdx.y;
    for (int idx = threadIdx.x; idx < 64 * KP; idx += 256) {
        int n = idx / KP, k = idx - n * KP;
        lw[idx] = (_Float16)W2[(size_t)(slice * 64 + n) * KP + k];
    }
    __syncthreads();

    int wave = threadIdx.x >> 5, lane = threadIdx.x & 31;
    int hi = lane >> 4, ln = lane & 15;
    int tile = blockIdx.x * 8 + wave;
    if (tile >= MTILES) return;

    int node = tile * 16 + ln;
    const float* hr = hin + (size_t)node * KIN;
    const float* sr = ssum + (size_t)node * DH;
    float rc = rcp[node];
    v16h A[KCH];
#pragma unroll
    for (int kc = 0; kc < KCH; ++kc) {
#pragma unroll
        for (int j = 0; j < 16; ++j) {
            int k = kc * 32 + a_k(j, hi);
            float v = (k < KIN) ? hr[k] : sr[k - KIN] * rc;
            A[kc][j] = (_Float16)v;
        }
    }
#pragma unroll
    for (int nt = 0; nt < 4; ++nt) {
        v8f acc = {};
        const _Float16* bb = &lw[(nt * 16 + ln) * KP];
#pragma unroll
        for (int kc = 0; kc < KCH; ++kc) {
            v16h B;
#pragma unroll
            for (int j = 0; j < 16; ++j) B[j] = bb[kc * 32 + hi * 16 + j];
            acc = wmma16x16x32(A[kc], B, acc);
        }
        int n = slice * 64 + nt * 16 + ln;
        float bv = b2[n];
#pragma unroll
        for (int v = 0; v < 8; ++v) {
            int rown = tile * 16 + v + 8 * hi;
            hout[(size_t)rown * DH + n] = fmaxf(acc[v] + bv, 0.f);
        }
    }
}

// ---------------------------------------------------------------------------
// Dense GEMM: Y = act(X @ W^T + b). X rows of length KP. grid.y slices of 64.
// ---------------------------------------------------------------------------
template <int KCH, bool RELU>
__global__ __launch_bounds__(256) void k_gemm(
    const float* __restrict__ X, const float* __restrict__ W,
    const float* __restrict__ bias, float* __restrict__ Y, int ldY) {
    constexpr int KP = KCH * 32;
    __shared__ _Float16 lw[64 * KP];
    int slice = blockIdx.y;
    for (int idx = threadIdx.x; idx < 64 * KP; idx += 256) {
        int n = idx / KP, k = idx - n * KP;
        lw[idx] = (_Float16)W[(size_t)(slice * 64 + n) * KP + k];
    }
    __syncthreads();

    int wave = threadIdx.x >> 5, lane = threadIdx.x & 31;
    int hi = lane >> 4, ln = lane & 15;
    int tile = blockIdx.x * 8 + wave;
    if (tile >= MTILES) return;

    const float* xr = X + (size_t)(tile * 16 + ln) * KP;
    v16h A[KCH];
#pragma unroll
    for (int kc = 0; kc < KCH; ++kc)
#pragma unroll
        for (int j = 0; j < 16; ++j)
            A[kc][j] = (_Float16)xr[kc * 32 + a_k(j, hi)];

#pragma unroll
    for (int nt = 0; nt < 4; ++nt) {
        v8f acc = {};
        const _Float16* bb = &lw[(nt * 16 + ln) * KP];
#pragma unroll
        for (int kc = 0; kc < KCH; ++kc) {
            v16h B;
#pragma unroll
            for (int j = 0; j < 16; ++j) B[j] = bb[kc * 32 + hi * 16 + j];
            acc = wmma16x16x32(A[kc], B, acc);
        }
        int n = slice * 64 + nt * 16 + ln;
        float bv = bias[n];
#pragma unroll
        for (int v = 0; v < 8; ++v) {
            float t = acc[v] + bv;
            if (RELU) t = fmaxf(t, 0.f);
            Y[(size_t)(tile * 16 + v + 8 * hi) * ldY + n] = t;
        }
    }
}

// Critic output: val[n] = dot(c1[n,:], Wc2) + bc2  (width-1, VALU is fine)
__global__ void k_values(const float* __restrict__ c1, const float* __restrict__ Wc2,
                         const float* __restrict__ bc2, float* __restrict__ out) {
    int i = blockIdx.x * blockDim.x + threadIdx.x;
    if (i < NNODES) {
        float s = bc2[0];
        const float* r = c1 + (size_t)i * 128;
#pragma unroll 4
        for (int k = 0; k < 128; ++k) s += r[k] * Wc2[k];
        out[i] = s;
    }
}

// ---------------------------------------------------------------------------
extern "C" void kernel_launch(void* const* d_in, const int* in_sizes, int n_in,
                              void* d_out, int out_size, void* d_ws, size_t ws_size,
                              hipStream_t stream) {
    (void)in_sizes; (void)n_in; (void)out_size; (void)ws_size;
    const int*   gate = (const int*)d_in[0];
    const int*   src  = (const int*)d_in[1];
    const int*   dst  = (const int*)d_in[2];
    const float* sidx = (const float*)d_in[3];
    const float* didx = (const float*)d_in[4];
    const float* rev  = (const float*)d_in[5];
    const float* emb  = (const float*)d_in[6];
    const float* W1_0 = (const float*)d_in[7];
    const float* W2_0 = (const float*)d_in[8];
    const float* b2_0 = (const float*)d_in[9];
    const float* W1s  = (const float*)d_in[10];
    const float* W2s  = (const float*)d_in[11];
    const float* b2s  = (const float*)d_in[12];
    const float* Wa1  = (const float*)d_in[13];
    const float* ba1  = (const float*)d_in[14];
    const float* Wa2  = (const float*)d_in[15];
    const float* ba2  = (const float*)d_in[16];
    const float* Wc1  = (const float*)d_in[17];
    const float* bc1  = (const float*)d_in[18];
    const float* Wc2  = (const float*)d_in[19];
    const float* bc2  = (const float*)d_in[20];

    float* out_actor = (float*)d_out;
    float* out_val   = out_actor + (size_t)NNODES * 256;

    // Workspace layout (floats), 418*N total (~167 MB), with overlays:
    //   [0,128N)      hB   (final h lives here)
    //   [128N,416N)   S    = h0(32N) | hA(128N) | ssum(128N); later c1 / a1
    //   [416N,418N)   deg, recip
    float*       ws  = (float*)d_ws;
    const size_t Nf  = NNODES;
    float* hB   = ws;
    float* S    = ws + 128 * Nf;
    float* h0   = S;
    float* hA   = S + 32 * Nf;
    float* ssum = S + 160 * Nf;
    float* deg  = ws + 416 * Nf;
    float* rcp  = ws + 417 * Nf;
    float* c1   = S;            // reuses S after convs
    float* a1   = S;            // reuses S after critic

    dim3 blk(256);
    const int edgeBlocks = NEDGES / 128;                 // 12500 (exact)
    const dim3 nodeGrid((MTILES + 7) / 8, 2);            // 782 x 2

    // Degrees (edge-structure invariant across layers)
    k_zero <<<(NNODES + 255) / 256, blk, 0, stream>>>(deg, NNODES);
    k_deg  <<<(NEDGES + 255) / 256, blk, 0, stream>>>(dst, deg);
    k_recip<<<(NNODES + 255) / 256, blk, 0, stream>>>(deg, rcp);

    // Embedding lookup
    k_embed<<<(NNODES * 32 + 255) / 256, blk, 0, stream>>>(gate, emb, h0);

    // conv_0: K = 32+3 -> pad 64 (2 chunks); W2 K = 160 (5 chunks)
    k_zero<<<(NNODES * DH + 255) / 256, blk, 0, stream>>>(ssum, NNODES * DH);
    k_edge<2, 32><<<edgeBlocks, blk, 0, stream>>>(h0, src, dst, sidx, didx, rev,
                                                  W1_0, ssum);
    k_node<5, 32><<<nodeGrid, blk, 0, stream>>>(h0, ssum, rcp, W2_0, b2_0, hA);

    // conv 1..5: K = 128+3 -> pad 160 (5 chunks); W2 K = 256 (8 chunks)
    for (int i = 0; i < 5; ++i) {
        const float* hin  = (i & 1) ? hB : hA;
        float*       hout = (i & 1) ? hA : hB;   // i=4 ends in hB
        k_zero<<<(NNODES * DH + 255) / 256, blk, 0, stream>>>(ssum, NNODES * DH);
        k_edge<5, 128><<<edgeBlocks, blk, 0, stream>>>(hin, src, dst, sidx, didx,
                                                       rev, W1s + (size_t)i * 128 * 131, ssum);
        k_node<8, 128><<<nodeGrid, blk, 0, stream>>>(hin, ssum, rcp,
                                                     W2s + (size_t)i * 128 * 256,
                                                     b2s + (size_t)i * 128, hout);
    }

    // Critic first (c1 overlays S, consumed before actor reuses S)
    k_gemm<4, true><<<dim3((MTILES + 7) / 8, 2), blk, 0, stream>>>(hB, Wc1, bc1, c1, 128);
    k_values<<<(NNODES + 255) / 256, blk, 0, stream>>>(c1, Wc2, bc2, out_val);

    // Actor
    k_gemm<4, true ><<<dim3((MTILES + 7) / 8, 4), blk, 0, stream>>>(hB, Wa1, ba1, a1, 256);
    k_gemm<8, false><<<dim3((MTILES + 7) / 8, 4), blk, 0, stream>>>(a1, Wa2, ba2, out_actor, 256);
}
